// S4DLayer_10393820856908
// MI455X (gfx1250) — matrix operand order repaired and checked by reading, not compiled, not run on hardware
//
#include <hip/hip_runtime.h>
#include <hip/hip_bf16.h>

// S4D layer for MI455X (gfx1250), FFT-free chunked-scan (SSD) formulation.
// B=8, H=512, L=4096, N=32 complex modes, chunk T=64, M=L/T=64 chunks.

typedef __attribute__((ext_vector_type(16))) __bf16 v16bf;
typedef __attribute__((ext_vector_type(8)))  float  v8f;
typedef int v4i_ __attribute__((vector_size(16)));   // 16-byte chunk type

#define B_  8
#define H_  512
#define L_  4096
#define N_  32
#define T_  64
#define M_  64
#define BL_ (B_ * L_)      // 32768 columns of the output GEMM
#define BM_ (B_ * M_)      // 512 chunk-columns per h

#if __has_builtin(__builtin_amdgcn_global_load_async_to_lds_b128)
#define HAVE_ASYNC_LDS 1
#endif

#ifdef HAVE_ASYNC_LDS
__device__ __forceinline__ void async_copy_b128(void* lds, const void* g) {
  __builtin_amdgcn_global_load_async_to_lds_b128(
      (__attribute__((address_space(1))) v4i_*)g,
      (__attribute__((address_space(3))) v4i_*)lds, 0, 0);
}
__device__ __forceinline__ void wait_async0() {
#if __has_builtin(__builtin_amdgcn_s_wait_asynccnt)
  __builtin_amdgcn_s_wait_asynccnt(0);
#else
  asm volatile("s_wait_asynccnt 0x0" ::: "memory");
#endif
}
#endif

// ---------------------------------------------------------------------------
// WMMA fragment layouts used here (wave32; contiguous k order, not the ISA table order):
//   A (16x32 bf16): lane m=lane&15, h2=lane>>4; elem e -> k_local =
//       16*h2 + e
//   B (32x16 bf16): n = lane&15, k_local = 16*h2 + e
//   C/D (16x16 f32): m = 8*h2 + r, n = lane&15
// LDS tiles are staged PRE-SWIZZLED in fragment order so each lane's 16
// elements are contiguous -> fragment load = 2x ds_load_b128, no packing.
// ---------------------------------------------------------------------------
__device__ __forceinline__ int afrag_idx(int m, int k, int mtiles) {
  int kstep = k >> 5;
  int kl = k & 31;
  int h2 = (kl >> 4) & 1;
  int e = kl & 15;
  int lane = (m & 15) | (h2 << 4);
  return ((kstep * mtiles + (m >> 4)) * 32 + lane) * 16 + e;
}

__device__ __forceinline__ int bfrag_idx(int k, int nlocal, int ntiles) {
  int kstep = k >> 5;
  int h2 = (k >> 4) & 1;
  int e = k & 15;
  int lane = (nlocal & 15) | (h2 << 4);
  return ((kstep * ntiles + (nlocal >> 4)) * 32 + lane) * 16 + e;
}

__device__ __forceinline__ v16bf load_afrag(const __bf16* sh, int kstep, int mt,
                                            int mtiles, int lane) {
  const __bf16* p = sh + ((kstep * mtiles + mt) * 32 + lane) * 16;
  v16bf f;
#pragma unroll
  for (int e = 0; e < 16; ++e) f[e] = p[e];
  return f;
}

__device__ __forceinline__ v16bf load_bfrag(const __bf16* sh, int kstep, int nt,
                                            int ntiles, int lane) {
  const __bf16* p = sh + ((kstep * ntiles + nt) * 32 + lane) * 16;
  v16bf f;
#pragma unroll
  for (int e = 0; e < 16; ++e) f[e] = p[e];
  return f;
}

__device__ __forceinline__ v8f wmma_bf16(v16bf a, v16bf b, v8f c) {
  return __builtin_amdgcn_wmma_f32_16x16x32_bf16(false, a, false, b,
                                                 (short)0, c, false, false);
}

// ---------------------------------------------------------------------------
// K1: per-(h) parameter kernel. 512 blocks x 64 threads (t = kernel tap).
// Produces: kloc[h][t] (local conv kernel), Ginter[h][t][2n] (inter-chunk
// output matrix), Wbar[h][n][j][2] = w^(63-j), wT[h][n][2] = w^64.
// ---------------------------------------------------------------------------
__global__ void s4d_param_kernel(const float* __restrict__ log_dt,
                                 const float* __restrict__ log_A_real,
                                 const float* __restrict__ A_imag,
                                 const float* __restrict__ Cv,
                                 float* __restrict__ kloc,
                                 float* __restrict__ Ginter,
                                 float* __restrict__ Wbar,
                                 float* __restrict__ wT) {
  int h = blockIdx.x;
  int t = threadIdx.x;  // 0..63
  float dt = expf(log_dt[h]);
  float ksum = 0.f;
  for (int n = 0; n < N_; ++n) {
    float Are = -expf(log_A_real[h * N_ + n]);
    float Aim = A_imag[h * N_ + n];
    float ar = Are * dt, ai = Aim * dt;            // dtA
    float er = expf(ar);
    float wr = er * cosf(ai), wi = er * sinf(ai);  // w = exp(dtA)
    // Cs = C * (w - 1) / A
    float cr = Cv[(h * N_ + n) * 2], ci = Cv[(h * N_ + n) * 2 + 1];
    float dr = wr - 1.f, di = wi;
    float inv = 1.f / (Are * Are + Aim * Aim);
    float qr = (dr * Are + di * Aim) * inv;
    float qi = (di * Are - dr * Aim) * inv;
    float csr = cr * qr - ci * qi;
    float csi = cr * qi + ci * qr;
    // w^t
    float tf = (float)t;
    float et = expf(ar * tf);
    float wtr = et * cosf(ai * tf), wti = et * sinf(ai * tf);
    ksum += 2.f * (csr * wtr - csi * wti);
    // Ginter row t: coefficients of (s_re, s_im) via Cs * w^(t+1)
    float w1r = wtr * wr - wti * wi, w1i = wtr * wi + wti * wr;
    Ginter[(h * T_ + t) * 64 + 2 * n]     =  2.f * (csr * w1r - csi * w1i);
    Ginter[(h * T_ + t) * 64 + 2 * n + 1] = -2.f * (csr * w1i + csi * w1r);
    // Wbar[j=t] = w^(63-t)
    float jf = (float)(T_ - 1 - t);
    float ej = expf(ar * jf);
    Wbar[((h * N_ + n) * T_ + t) * 2]     = ej * cosf(ai * jf);
    Wbar[((h * N_ + n) * T_ + t) * 2 + 1] = ej * sinf(ai * jf);
  }
  kloc[h * T_ + t] = ksum;
  if (t < N_) {  // w^T for the chunk scan
    int n = t;
    float Are = -expf(log_A_real[h * N_ + n]);
    float Aim = A_imag[h * N_ + n];
    float ar = Are * dt * (float)T_, ai = Aim * dt * (float)T_;
    float e = expf(ar);
    wT[(h * N_ + n) * 2]     = e * cosf(ai);
    wT[(h * N_ + n) * 2 + 1] = e * sinf(ai);
  }
}

// ---------------------------------------------------------------------------
// K2: per-(b,h) chunk projection q[c,n] = sum_j w^(63-j) u[cT+j], then a
// 64-step scan s[c+1] = w^T s[c] + q[c]. Writes chunk-start states
// S[b][h][c][2n{re},2n+1{im}]. 4096 blocks x 128 threads.
// Staging uses CDNA5 async global->LDS copies when the toolchain has them.
// ---------------------------------------------------------------------------
__global__ void s4d_qscan_kernel(const float* __restrict__ u,
                                 const float* __restrict__ Wbar,
                                 const float* __restrict__ wT,
                                 float* __restrict__ S) {
  __shared__ __align__(16) float u_sh[L_];     // 16KB
  __shared__ __align__(16) float wb_sh[4096];  // 16KB  [n][j][2]
  __shared__ __align__(16) float q_sh[4096];   // 16KB  [c][n][2]
  int tid = threadIdx.x;
  int b = blockIdx.x >> 9, h = blockIdx.x & (H_ - 1);
  const float* ub = u + ((size_t)(b * H_ + h)) * L_;
  const float* wbh = Wbar + (size_t)h * 4096;
#ifdef HAVE_ASYNC_LDS
#pragma unroll
  for (int it = 0; it < 8; ++it) {
    int off = (it * 128 + tid) * 4;  // 16-byte chunks
    async_copy_b128(&u_sh[off], &ub[off]);
    async_copy_b128(&wb_sh[off], &wbh[off]);
  }
  wait_async0();
#else
  for (int i = tid; i < L_; i += 128) u_sh[i] = ub[i];
  for (int i = tid; i < 4096; i += 128) wb_sh[i] = wbh[i];
#endif
  __syncthreads();
  for (int p = 0; p < 16; ++p) {
    int idx = tid + p * 128;          // 0..2047 -> (c,n)
    int c = idx >> 5, n = idx & 31;
    const float* wn = wb_sh + n * 128;
    const float* uc = u_sh + c * T_;
    float qr = 0.f, qi = 0.f;
    for (int j = 0; j < T_; ++j) {
      float uv = uc[j];
      qr += wn[2 * j] * uv;
      qi += wn[2 * j + 1] * uv;
    }
    q_sh[idx * 2] = qr;
    q_sh[idx * 2 + 1] = qi;
  }
  __syncthreads();
  if (tid < N_) {
    int n = tid;
    float sr = 0.f, si = 0.f;
    float wr = wT[(h * N_ + n) * 2], wi = wT[(h * N_ + n) * 2 + 1];
    float* Sb = S + ((size_t)(b * H_ + h)) * M_ * 64;
    for (int c = 0; c < M_; ++c) {
      Sb[c * 64 + 2 * n] = sr;
      Sb[c * 64 + 2 * n + 1] = si;
      float qr = q_sh[(c * 32 + n) * 2], qi = q_sh[(c * 32 + n) * 2 + 1];
      float nsr = wr * sr - wi * si + qr;
      si = wr * si + wi * sr + qi;
      sr = nsr;
    }
  }
}

// ---------------------------------------------------------------------------
// K3: y = Toeplitz(k) @ U  +  Ginter @ S  + D*u, then exact GELU; store bf16
// activations Gact[h][b][l]. Grid (32 col-tiles, 512 h) x 128 threads;
// each of 4 waves owns one 16-row M-tile of the 64x16 output tile. WMMA.
// All LDS tiles staged pre-swizzled in fragment order.
// ---------------------------------------------------------------------------
__global__ void s4d_combine_kernel(const float* __restrict__ u,
                                   const float* __restrict__ kloc,
                                   const float* __restrict__ Ginter,
                                   const float* __restrict__ S,
                                   const float* __restrict__ Dp,
                                   __bf16* __restrict__ Gact) {
  __shared__ __align__(32) __bf16 kmat_sh[T_ * T_];   // 8KB  A-frag order
  __shared__ __align__(32) __bf16 gint_sh[T_ * 64];   // 8KB  A-frag order
  __shared__ __align__(32) __bf16 ub_sh[T_ * 16];     // 2KB  B-frag order
  __shared__ __align__(32) __bf16 sb_sh[64 * 16];     // 2KB  B-frag order
  int tid = threadIdx.x, lane = tid & 31, wv = tid >> 5;
  int h = blockIdx.y;
  int colbase = blockIdx.x * 16;        // col = b*64 + c
  const float* kh = kloc + h * T_;
  for (int i = tid; i < T_ * T_; i += 128) {
    int t = i >> 6, j = i & 63;
    kmat_sh[afrag_idx(t, j, 4)] = (__bf16)((j <= t) ? kh[t - j] : 0.f);
  }
  const float* gh = Ginter + (size_t)h * 4096;
  for (int i = tid; i < 4096; i += 128) {
    int t = i >> 6, k2 = i & 63;
    gint_sh[afrag_idx(t, k2, 4)] = (__bf16)gh[i];
  }
  for (int i = tid; i < 1024; i += 128) {
    int j = i >> 4, cl = i & 15;
    int col = colbase + cl;
    int b = col >> 6, c = col & 63;
    ub_sh[bfrag_idx(j, cl, 1)] =
        (__bf16)u[((size_t)(b * H_ + h)) * L_ + c * T_ + j];
    sb_sh[bfrag_idx(j, cl, 1)] =
        (__bf16)S[(((size_t)(b * H_ + h)) * M_ + c) * 64 + j];
  }
  __syncthreads();
  v8f acc = {};
#pragma unroll
  for (int ks = 0; ks < 2; ++ks) {
    v16bf a1 = load_afrag(kmat_sh, ks, wv, 4, lane);
    v16bf b1 = load_bfrag(ub_sh, ks, 0, 1, lane);
    acc = wmma_bf16(a1, b1, acc);
  }
#pragma unroll
  for (int ks = 0; ks < 2; ++ks) {
    v16bf a2 = load_afrag(gint_sh, ks, wv, 4, lane);
    v16bf b2 = load_bfrag(sb_sh, ks, 0, 1, lane);
    acc = wmma_bf16(a2, b2, acc);
  }
  float Dh = Dp[h];
  int cl = lane & 15;
  int col = colbase + cl;
  int b = col >> 6, c = col & 63;
  const float* ubase = u + ((size_t)(b * H_ + h)) * L_ + c * T_;
  __bf16* gbase = Gact + (size_t)h * BL_ + b * L_ + c * T_;
#pragma unroll
  for (int r = 0; r < 8; ++r) {
    int t = wv * 16 + ((lane >> 4) << 3) + r;
    float y = acc[r] + Dh * ubase[t];
    y = 0.5f * y * (1.f + erff(y * 0.70710678118f));  // exact GELU
    gbase[t] = (__bf16)y;
  }
}

// ---------------------------------------------------------------------------
// K4: 1x1 conv GEMM (1024 x 512) @ (512 x 32768) with fused GLU.
// Grid (1024 col-blocks of 32, 8 row-blocks of 64) x 128 threads.
// Wave wv computes rows o=oblock+16wv (a-half) and o+512 (b-half) for two
// 16-col tiles; out = a * sigmoid(b). 256 WMMAs per block; LDS pre-swizzled.
// ---------------------------------------------------------------------------
__global__ void s4d_outproj_kernel(const __bf16* __restrict__ Gact,
                                   const float* __restrict__ W,
                                   const float* __restrict__ bo,
                                   float* __restrict__ out) {
  __shared__ __align__(32) __bf16 gb_sh[512 * 32];  // 32KB  B-frag order
  __shared__ __align__(32) __bf16 wa_sh[128 * 32];  // 8KB   A-frag order
  int tid = threadIdx.x, lane = tid & 31, wv = tid >> 5;
  int colbase = blockIdx.x * 32;       // col = b*4096 + l
  int oblock = blockIdx.y * 64;
  for (int i = tid; i < 512 * 32; i += 128) {
    int k = i >> 5, cl = i & 31;
    gb_sh[bfrag_idx(k, cl, 2)] = Gact[(size_t)k * BL_ + colbase + cl];
  }
  v8f accA0 = {}, accA1 = {}, accB0 = {}, accB1 = {};
  for (int ks = 0; ks < 16; ++ks) {
    __syncthreads();
    for (int i = tid; i < 128 * 32; i += 128) {
      int r = i >> 5, kk = i & 31;
      int o = (r < 64) ? (oblock + r) : (512 + oblock + (r - 64));
      wa_sh[afrag_idx(r, kk, 8)] = (__bf16)W[(size_t)o * H_ + ks * 32 + kk];
    }
    if (ks + 1 < 16) {  // warm next W tile in L2 (global_prefetch_b8)
      int r = tid;
      int o = (r < 64) ? (oblock + r) : (512 + oblock + (r - 64));
      __builtin_prefetch(&W[(size_t)o * H_ + (ks + 1) * 32], 0, 0);
    }
    __syncthreads();
    v16bf aA = load_afrag(wa_sh, 0, wv, 8, lane);
    v16bf aB = load_afrag(wa_sh, 0, 4 + wv, 8, lane);
    v16bf b0 = load_bfrag(gb_sh, ks, 0, 2, lane);
    v16bf b1 = load_bfrag(gb_sh, ks, 1, 2, lane);
    accA0 = wmma_bf16(aA, b0, accA0);
    accA1 = wmma_bf16(aA, b1, accA1);
    accB0 = wmma_bf16(aB, b0, accB0);
    accB1 = wmma_bf16(aB, b1, accB1);
  }
#pragma unroll
  for (int nt = 0; nt < 2; ++nt) {
    v8f av8 = nt ? accA1 : accA0;
    v8f bv8 = nt ? accB1 : accB0;
    int col = colbase + nt * 16 + (lane & 15);
    int b = col >> 12, l = col & 4095;
#pragma unroll
    for (int r = 0; r < 8; ++r) {
      int o = oblock + wv * 16 + ((lane >> 4) << 3) + r;
      float a = av8[r] + bo[o];
      float g = bv8[r] + bo[o + 512];
      out[((size_t)(b * H_ + o)) * L_ + l] = a * (1.f / (1.f + expf(-g)));
    }
  }
}

// ---------------------------------------------------------------------------
extern "C" void kernel_launch(void* const* d_in, const int* in_sizes, int n_in,
                              void* d_out, int out_size, void* d_ws, size_t ws_size,
                              hipStream_t stream) {
  (void)in_sizes; (void)n_in; (void)out_size; (void)ws_size;
  const float* u          = (const float*)d_in[0];
  const float* log_dt     = (const float*)d_in[1];
  const float* log_A_real = (const float*)d_in[2];
  const float* A_imag     = (const float*)d_in[3];
  const float* Cv         = (const float*)d_in[4];
  const float* D          = (const float*)d_in[5];
  const float* W_out      = (const float*)d_in[6];
  const float* b_out      = (const float*)d_in[7];
  float* out = (float*)d_out;

  // workspace layout (f32 unless noted): ~118 MB total
  float* kloc   = (float*)d_ws;                 // H*T                = 32768
  float* Ginter = kloc + 32768;                 // H*T*64             = 2097152
  float* Wbar   = Ginter + 2097152;             // H*N*T*2            = 2097152
  float* wT     = Wbar + 2097152;               // H*N*2              = 32768
  float* S      = wT + 32768;                   // B*H*M*64           = 16777216
  __bf16* Gact  = (__bf16*)(S + 16777216);      // H*B*L bf16         = 16777216

  s4d_param_kernel<<<H_, T_, 0, stream>>>(log_dt, log_A_real, A_imag, Cv,
                                          kloc, Ginter, Wbar, wT);
  s4d_qscan_kernel<<<B_ * H_, 128, 0, stream>>>(u, Wbar, wT, S);
  s4d_combine_kernel<<<dim3(BM_ / 16, H_), 128, 0, stream>>>(u, kloc, Ginter,
                                                             S, D, Gact);
  s4d_outproj_kernel<<<dim3(BL_ / 32, H_ / 64), 128, 0, stream>>>(Gact, W_out,
                                                                  b_out, out);
}
